// DeepMatrixtModel_45011257262087
// MI455X (gfx1250) — matrix operand order repaired and checked by reading, not compiled, not run on hardware
//
#include <hip/hip_runtime.h>
#include <hip/hip_bf16.h>

typedef __attribute__((ext_vector_type(16))) _Float16 v16h;
typedef __attribute__((ext_vector_type(8)))  float    v8f;

#define KDIM 64

union H2 { unsigned int u; _Float16 h[2]; };

// sigmoid via v_exp_f32 + v_rcp_f32 (no IEEE div_scale/div_fixup chain;
// 1-ULP rcp error is far below the f16 WMMA quantization already accepted)
__device__ __forceinline__ float fast_sigmoid(float x) {
    return __builtin_amdgcn_rcpf(1.0f + __expf(-x));
}

// ---------------------------------------------------------------------------
// Kernel 0: zero the per-user accumulator + count table in workspace
// ---------------------------------------------------------------------------
__global__ void zero_f32_kernel(float* p, long long n) {
    long long i = (long long)blockIdx.x * blockDim.x + threadIdx.x;
    if (i < n) p[i] = 0.0f;
}

// ---------------------------------------------------------------------------
// Kernel 1: rbm = sigmoid((feat[movie] * sigmoid(r/5 * W_r + b_r)) @ W_c^T + b_c)
//           scatter-add into per-user accumulator; count ratings per user;
//           persist rbm rows < NB_MOVIES for the finalize pass.
// One wave (wave32) owns a 16-row tile; 256-thread block = 8 waves = 128 rows.
// GEMM: v_wmma_f32_16x16x32_f16, two K-slices chained on one f32 accumulator.
// ---------------------------------------------------------------------------
__global__ void __launch_bounds__(256)
rbm_scatter_kernel(const int* __restrict__ data_users,
                   const int* __restrict__ data_movies,
                   const float* __restrict__ data_ratings,
                   const float* __restrict__ feat,      // [NB_MOVIES][64]
                   const float* __restrict__ W_r,       // [64]
                   const float* __restrict__ b_r,       // [64]
                   const float* __restrict__ W_c,       // [64][64] row-major (out j, in k)
                   const float* __restrict__ b_c,       // [64]
                   float* __restrict__ rbu,             // [U][64] accumulator
                   float* __restrict__ nbcnt,           // [U]
                   float* __restrict__ rbm_store,       // [NB_MOVIES][64]
                   int N, int nb_movies)
{
    __shared__ _Float16 s_wc[KDIM * KDIM];   // s_wc[n*64+k] = W_c[n][k] (f16)
    __shared__ float    s_wr[KDIM];
    __shared__ float    s_br[KDIM];
    __shared__ float    s_bc[KDIM];

    const int tid = threadIdx.x;
    for (int i = tid; i < KDIM * KDIM; i += 256) s_wc[i] = (_Float16)W_c[i];
    if (tid < KDIM) { s_wr[tid] = W_r[tid]; s_br[tid] = b_r[tid]; s_bc[tid] = b_c[tid]; }
    __syncthreads();

    const int wave = tid >> 5;
    const int lane = tid & 31;
    const int tile_base = (blockIdx.x * 8 + wave) * 16;
    if (tile_base >= N) return;   // whole-wave uniform exit; WMMA waves keep EXEC all-ones

    const bool lowHalf   = (lane < 16);
    const bool full_tile = (tile_base + 16 <= N);           // wave-uniform
    const bool may_store = (tile_base < nb_movies);         // wave-uniform
    const int  M    = lane & 15;
    const int  row  = tile_base + M;
    const int  rowc = row < N ? row : (N - 1);

    const int   movie = data_movies[rowc];
    const int   user  = data_users[rowc];
    const float rn    = data_ratings[rowc] * 0.2f;   // /MAX_RATING

    // ---- Build A (16x32 f16 per WMMA) for both K slices --------------------
    // lane holds row M; K pairs: j<4 -> k0+2j ; j>=4 -> k0+16+2(j-4); k0 = ks*32 + (lowHalf?0:8)
    v16h a0, a1;
    {
        const int kbase = lowHalf ? 0 : 8;
        #pragma unroll
        for (int ks = 0; ks < 2; ++ks) {
            const int k0 = ks * 32 + kbase;
            #pragma unroll
            for (int j = 0; j < 8; ++j) {
                const int k = k0 + ((j < 4) ? (2 * j) : (16 + 2 * (j - 4)));
                const float2 f = *(const float2*)&feat[movie * KDIM + k];
                const float e0 = fast_sigmoid(rn * s_wr[k]     + s_br[k]);
                const float e1 = fast_sigmoid(rn * s_wr[k + 1] + s_br[k + 1]);
                const _Float16 h0 = (_Float16)(f.x * e0);
                const _Float16 h1 = (_Float16)(f.y * e1);
                if (ks == 0) { a0[2 * j] = h0; a0[2 * j + 1] = h1; }
                else         { a1[2 * j] = h0; a1[2 * j + 1] = h1; }
            }
        }
    }

    // ---- Gather users of the 8 D-rows this lane owns (no shuffles) ---------
    // D layout: lane owns rows M = v + (lowHalf?0:8), column N = lane&15.
    const int ncol = lane & 15;
    const int off8 = lowHalf ? 0 : 8;

    float* ap[8];   // accumulator base addrs; per-ct offset folds into the atomic's imm
    float* sp = rbm_store + (long long)(tile_base + off8) * KDIM + ncol;
    if (full_tile) {
        const int4 ua = *(const int4*)&data_users[tile_base + off8];
        const int4 ub = *(const int4*)&data_users[tile_base + off8 + 4];
        ap[0] = rbu + (long long)ua.x * KDIM + ncol;
        ap[1] = rbu + (long long)ua.y * KDIM + ncol;
        ap[2] = rbu + (long long)ua.z * KDIM + ncol;
        ap[3] = rbu + (long long)ua.w * KDIM + ncol;
        ap[4] = rbu + (long long)ub.x * KDIM + ncol;
        ap[5] = rbu + (long long)ub.y * KDIM + ncol;
        ap[6] = rbu + (long long)ub.z * KDIM + ncol;
        ap[7] = rbu + (long long)ub.w * KDIM + ncol;
    } else {
        #pragma unroll
        for (int v = 0; v < 8; ++v) {
            const int grow = tile_base + off8 + v;
            const int u    = data_users[grow < N ? grow : (N - 1)];
            ap[v] = rbu + (long long)u * KDIM + ncol;
        }
    }

    // ---- 4 column tiles of 16, K=64 via two chained WMMAs ------------------
    const int kbB = lowHalf ? 0 : 8;
    #pragma unroll
    for (int ct = 0; ct < 4; ++ct) {
        const int col = ct * 16 + ncol;

        v8f c;
        const float binit = s_bc[col];
        #pragma unroll
        for (int v = 0; v < 8; ++v) c[v] = binit;   // bias rides the accumulator

        // B (32x16 f16): VGPR j holds K pair: j<4 -> kb+2j ; j>=4 -> kb+16+2(j-4), for column `col`
        v16h b0, b1;
        #pragma unroll
        for (int j = 0; j < 8; ++j) {
            H2 w;
            w.u = *(const unsigned int*)&s_wc[col * KDIM + kbB + ((j < 4) ? (2 * j) : (16 + 2 * (j - 4)))];
            b0[2 * j] = w.h[0]; b0[2 * j + 1] = w.h[1];
            w.u = *(const unsigned int*)&s_wc[col * KDIM + 32 + kbB + ((j < 4) ? (2 * j) : (16 + 2 * (j - 4)))];
            b1[2 * j] = w.h[0]; b1[2 * j + 1] = w.h[1];
        }

        c = __builtin_amdgcn_wmma_f32_16x16x32_f16(false, a0, false, b0, (short)0, c, false, false);
        c = __builtin_amdgcn_wmma_f32_16x16x32_f16(false, a1, false, b1, (short)0, c, false, false);

        // sigmoid + scatter; ct*16 floats = ct*64 bytes folds into the imm offset
        if (full_tile) {
            #pragma unroll
            for (int v = 0; v < 8; ++v) {
                const float val = fast_sigmoid(c[v]);
                atomicAdd(ap[v] + ct * 16, val);
                if (may_store) {
                    if (tile_base + off8 + v < nb_movies)
                        sp[v * KDIM + ct * 16] = val;
                }
            }
        } else {            // cold tail (unused when 128 | N)
            #pragma unroll
            for (int v = 0; v < 8; ++v) {
                const float val  = fast_sigmoid(c[v]);
                const int   grow = tile_base + off8 + v;
                if (grow < N) {
                    atomicAdd(ap[v] + ct * 16, val);
                    if (grow < nb_movies)
                        sp[v * KDIM + ct * 16] = val;
                }
            }
        }
    }

    // ---- per-row count (one lane per row) ----------------------------------
    if (lowHalf && row < N)
        atomicAdd(&nbcnt[user], 1.0f);
}

// ---------------------------------------------------------------------------
// Kernel 2: per-query finalize (x_users is identity in setup, handled faithfully)
// ---------------------------------------------------------------------------
__global__ void finalize_kernel(const int* __restrict__ x_users,
                                const int* __restrict__ x_movies,
                                const float* __restrict__ feat,
                                const float* __restrict__ rbu,
                                const float* __restrict__ nbcnt,
                                const float* __restrict__ rbm_store,
                                const float* __restrict__ w_out,
                                const float* __restrict__ b_out,
                                float* __restrict__ out, int B)
{
    const int i = blockIdx.x * blockDim.x + threadIdx.x;
    if (i >= B) return;
    const int u = x_users[i];
    const int m = x_movies[i];

    const float nb  = fmaxf(1.0f, nbcnt[u] - 1.0f);
    const float inv = 1.0f / nb;

    const float4* ru = (const float4*)&rbu[(long long)u * KDIM];
    const float4* rs = (const float4*)&rbm_store[(long long)m * KDIM];
    const float4* mf = (const float4*)&feat[(long long)m * KDIM];

    float dot = 0.0f;
    #pragma unroll
    for (int q = 0; q < KDIM / 4; ++q) {
        const float4 a = ru[q];
        const float4 s = rs[q];
        const float4 f = mf[q];
        dot += f.x * (a.x - s.x) + f.y * (a.y - s.y)
             + f.z * (a.z - s.z) + f.w * (a.w - s.w);
    }
    dot *= inv;
    const float r = dot * (1.0f / (float)KDIM);
    out[i] = (r * w_out[0] + b_out[0]) * 5.0f;
}

// ---------------------------------------------------------------------------
extern "C" void kernel_launch(void* const* d_in, const int* in_sizes, int n_in,
                              void* d_out, int out_size, void* d_ws, size_t ws_size,
                              hipStream_t stream)
{
    const int*   x_users      = (const int*)  d_in[0];
    const int*   x_movies     = (const int*)  d_in[1];
    const int*   data_users   = (const int*)  d_in[2];
    const int*   data_movies  = (const int*)  d_in[3];
    const float* data_ratings = (const float*)d_in[4];
    const float* feat         = (const float*)d_in[5];
    const float* W_r          = (const float*)d_in[6];
    const float* b_r          = (const float*)d_in[7];
    const float* W_c          = (const float*)d_in[8];
    const float* b_c          = (const float*)d_in[9];
    const float* w_out        = (const float*)d_in[10];
    const float* b_out        = (const float*)d_in[11];
    float*       out          = (float*)d_out;

    const int U   = in_sizes[0];           // 100000 users queried
    const int N   = in_sizes[2];           // 2,000,000 data points
    const int NBM = in_sizes[5] / KDIM;    // 20,000 movies

    // workspace layout
    float* rbu       = (float*)d_ws;               // U*64
    float* nbcnt     = rbu + (long long)U * KDIM;  // U
    float* rbm_store = nbcnt + U;                  // NBM*64

    // zero accumulators (rbu + nbcnt)
    {
        long long nz = (long long)U * KDIM + U;
        int blocks = (int)((nz + 255) / 256);
        zero_f32_kernel<<<blocks, 256, 0, stream>>>(rbu, nz);
    }

    // main WMMA + scatter pass: 128 rows per 256-thread block
    {
        int blocks = (N + 127) / 128;
        rbm_scatter_kernel<<<blocks, 256, 0, stream>>>(
            data_users, data_movies, data_ratings, feat,
            W_r, b_r, W_c, b_c,
            rbu, nbcnt, rbm_store, N, NBM);
    }

    // finalize per query
    {
        int blocks = (U + 255) / 256;
        finalize_kernel<<<blocks, 256, 0, stream>>>(
            x_users, x_movies, feat, rbu, nbcnt, rbm_store,
            w_out, b_out, out, U);
    }
}
